// TSB_18305150616316
// MI455X (gfx1250) — matrix operand order repaired and checked by reading, not compiled, not run on hardware
//
#include <hip/hip_runtime.h>
#include <hip/hip_bf16.h>

// ---------------- problem constants (from reference setup_inputs) ----------
#define BB 16
#define CC 2048
#define TT 16
#define HH 16
#define WW 8
#define HWV 128           // h*w
#define NN 2048           // t*h*w
#define EPS_NORM 1e-12f
#define EPS_BN 1e-5f
#define SCALEF 5.0f
#define MASK_VAL 1e8f

#define CSEG 8            // channel segments for k_pool (blocks per (b,t))
#define CPART (CC / CSEG) // 256 channels per block
#define CCHUNK 64         // channels per LDS tile
#define TSTRIDE (CCHUNK + 16) // 80 halfs -> 160B row stride (16B aligned)

typedef __attribute__((ext_vector_type(16))) _Float16 v16h;
typedef __attribute__((ext_vector_type(8)))  _Float16 v8h;
typedef __attribute__((ext_vector_type(8)))  float    v8f;

// ---------------------------------------------------------------------------
// Fragment helpers (contiguous k order used by this code, wave32)
//   A 16x32 f16 : lane l -> M = l&15, g = l>>4 ; half i at K = g*16+i
//   B 32x16 f16 : lane l -> Ncol = l&15, g = l>>4 ; half i at K = g*16+i
// ---------------------------------------------------------------------------
__device__ __forceinline__ v16h load_fragA(const _Float16* row, int k0, int g) {
    v8h lo = *(const v8h*)(row + k0 + g * 16);
    v8h hi = *(const v8h*)(row + k0 + g * 16 + 8);
    return __builtin_shufflevector(lo, hi, 0,1,2,3,4,5,6,7,8,9,10,11,12,13,14,15);
}

__device__ __forceinline__ v16h load_fragB_h(const _Float16* colbase, int k0) {
    return *(const v16h*)(colbase + k0);   // colbase already includes g*16
}

__device__ __forceinline__ v16h load_fragB_f32(const float* colbase, int k0) {
    v16h r;
    const float4* p = (const float4*)(colbase + k0);
#pragma unroll
    for (int i = 0; i < 4; ++i) {
        float4 v = p[i];
        r[4*i+0] = (_Float16)v.x;
        r[4*i+1] = (_Float16)v.y;
        r[4*i+2] = (_Float16)v.z;
        r[4*i+3] = (_Float16)v.w;
    }
    return r;
}

// ---------------------------------------------------------------------------
// K1: 2x2 avg-pool (pad 1, crop) -> mem staged f16 as (b, n, c) via LDS
//     transpose so stores are coalesced 64B bursts.  Also emits per-segment
//     partial column sum-of-squares (from the f16-rounded values).
//     grid = (B*T, CSEG), block = 256.
// ---------------------------------------------------------------------------
__global__ void k_pool(const float* __restrict__ x,
                       _Float16* __restrict__ mn,
                       float* __restrict__ norm2m_part) {
    const int bt  = blockIdx.x;
    const int b   = bt / TT, t = bt % TT;
    const int c0  = blockIdx.y * CPART;
    const int tid = threadIdx.x;
    const int row = tid >> 1;          // pixel row this thread writes out
    const int hsel = tid & 1;          // which 32-channel half of the chunk

    __shared__ _Float16 tile[HWV][TSTRIDE];
    __shared__ float ssbuf[256];

    const float* base = x + (((size_t)b * CC) * TT + t) * HWV;
    float ss = 0.f;

    for (int cc = 0; cc < CPART; cc += CCHUNK) {
        // ---- compute pooled tile: CCHUNK channels x 128 pixels ----
#pragma unroll
        for (int it = 0; it < (CCHUNK * HWV) / 256; ++it) {
            int idx = it * 256 + tid;
            int ci = idx >> 7, p = idx & 127;
            int c = c0 + cc + ci;
            const float* xb = base + (size_t)c * TT * HWV;
            int i = p >> 3, j = p & 7;
            float ctr = xb[p];
            float l  = (j > 0) ? xb[p - 1] : 0.f;
            float u  = (i > 0) ? xb[p - 8] : 0.f;
            float ul = (i > 0 && j > 0) ? xb[p - 9] : 0.f;
            tile[p][ci] = (_Float16)(0.25f * (ctr + l + u + ul));
        }
        __syncthreads();
        // ---- transpose-write: 2 threads per pixel row, 32 halfs (64B) each --
        {
            _Float16* orow = mn + ((size_t)b * NN + (size_t)t * HWV + row) * CC
                             + c0 + cc + hsel * 32;
            const _Float16* src = &tile[row][hsel * 32];
#pragma unroll
            for (int q = 0; q < 32; ++q) { float v = (float)src[q]; ss += v * v; }
#pragma unroll
            for (int q = 0; q < 4; ++q)
                ((uint4*)orow)[q] = ((const uint4*)src)[q];
        }
        __syncthreads();
    }

    ssbuf[tid] = ss;
    __syncthreads();
    if (hsel == 0) {
        float tot = ssbuf[tid] + ssbuf[tid + 1];
        norm2m_part[((size_t)blockIdx.y * BB + b) * NN + (size_t)t * HWV + row] = tot;
    }
}

// ---------------------------------------------------------------------------
// K2: query = spatial mean -> qn_h (b, t, c) f16.  One wave per (b,t,c) row.
// ---------------------------------------------------------------------------
__global__ void k_query(const float* __restrict__ x, _Float16* __restrict__ qn) {
    const int gid  = blockIdx.x * 8 + (threadIdx.x >> 5);
    const int lane = threadIdx.x & 31;
    const int c  = gid % CC;
    const int bt = gid / CC;
    const int t  = bt % TT, b = bt / TT;

    const float4 v = ((const float4*)(x + (((size_t)b * CC + c) * TT + t) * HWV))[lane];
    float s = v.x + v.y + v.z + v.w;
#pragma unroll
    for (int m = 16; m > 0; m >>= 1) s += __shfl_xor(s, m, 32);
    if (lane == 0)
        qn[((size_t)b * TT + t) * CC + c] = (_Float16)(s * (1.0f / 128.0f));
}

// ---------------------------------------------------------------------------
// K3: raw score GEMM per b:  S[t, n] = sum_c q[t,c] * m[n,c]   (f16 WMMA)
//     grid = B * (N/128), block = 256 (8 waves, 16 cols each).
// ---------------------------------------------------------------------------
__global__ void k_score(const _Float16* __restrict__ qn,
                        const _Float16* __restrict__ mn,
                        float* __restrict__ scores) {
    const int wave = threadIdx.x >> 5;
    const int lane = threadIdx.x & 31;
    const int b    = blockIdx.x / (NN / 128);
    const int col0 = (blockIdx.x % (NN / 128)) * 128 + wave * 16;
    const int m    = lane & 15, g = lane >> 4;

    const _Float16* arow = qn + ((size_t)b * TT + m) * CC;
    const _Float16* bcol = mn + ((size_t)b * NN + col0 + m) * CC + g * 16;

    v8f acc = {};
    for (int k0 = 0; k0 < CC; k0 += 32) {
        __builtin_prefetch(arow + k0 + 128, 0, 0);
        __builtin_prefetch(bcol + k0 + 128, 0, 0);
        v16h a  = load_fragA(arow, k0, g);
        v16h bf = load_fragB_h(bcol, k0);
        acc = __builtin_amdgcn_wmma_f32_16x16x32_f16(false, a, false, bf,
                                                     (short)0, acc, false, false);
    }
    float* out = scores + (size_t)b * TT * NN;
#pragma unroll
    for (int r = 0; r < 8; ++r) {
        int row = r + g * 8;                         // D: lane -> N=l&15, VGPR r -> M
        out[(size_t)row * NN + col0 + m] = acc[r];
    }
}

// ---------------------------------------------------------------------------
// K4: fused q-norm + cosine rescale + diag mask + softmax -> probs f16.
//     grid = B*T, block = 256 (each thread covers 8 of 2048 columns).
// ---------------------------------------------------------------------------
__global__ void k_softmax(const float* __restrict__ scores,
                          const _Float16* __restrict__ qn,
                          const float* __restrict__ norm2m_part,
                          _Float16* __restrict__ probs) {
    const int bt = blockIdx.x;
    const int b = bt / TT, t = bt % TT;
    const int tid = threadIdx.x;
    __shared__ float red[256];

    // ||q||^2 from the f16 values actually used in the GEMM
    const _Float16* qrow = qn + (size_t)bt * CC;
    float sq = 0.f;
    for (int i = tid; i < CC; i += 256) { float v = (float)qrow[i]; sq += v * v; }
    red[tid] = sq; __syncthreads();
    for (int s = 128; s > 0; s >>= 1) { if (tid < s) red[tid] += red[tid + s]; __syncthreads(); }
    const float invq = 1.f / fmaxf(sqrtf(red[0]), EPS_NORM);
    __syncthreads();

    const float* srow = scores + (size_t)bt * NN;
    float sv[8];
    float mx = -1e30f;
#pragma unroll
    for (int k = 0; k < 8; ++k) {
        int n = k * 256 + tid;
        float n2 = 0.f;
#pragma unroll
        for (int s = 0; s < CSEG; ++s)
            n2 += norm2m_part[((size_t)s * BB + b) * NN + n];
        float invm = 1.f / fmaxf(sqrtf(n2), EPS_NORM);
        float s = srow[n] * invq * invm * SCALEF;
        if ((n >> 7) == t) s -= MASK_VAL;            // mask own frame
        sv[k] = s; mx = fmaxf(mx, s);
    }
    red[tid] = mx; __syncthreads();
    for (int s = 128; s > 0; s >>= 1) { if (tid < s) red[tid] = fmaxf(red[tid], red[tid + s]); __syncthreads(); }
    const float gmax = red[0]; __syncthreads();

    float se = 0.f;
#pragma unroll
    for (int k = 0; k < 8; ++k) { float e = __expf(sv[k] - gmax); sv[k] = e; se += e; }
    red[tid] = se; __syncthreads();
    for (int s = 128; s > 0; s >>= 1) { if (tid < s) red[tid] += red[tid + s]; __syncthreads(); }
    const float inv = 1.f / red[0];

    _Float16* prow = probs + (size_t)bt * NN;
#pragma unroll
    for (int k = 0; k < 8; ++k)
        prow[k * 256 + tid] = (_Float16)(sv[k] * inv);
}

// ---------------------------------------------------------------------------
// K5: aggregation GEMM per b: Y[t, c] = sum_n P[t,n] * x[c,n]  (x f32->f16)
//     grid = B * (C/128), block = 256.
// ---------------------------------------------------------------------------
__global__ void k_agg(const _Float16* __restrict__ probs,
                      const float* __restrict__ x,
                      _Float16* __restrict__ yh) {
    const int wave = threadIdx.x >> 5;
    const int lane = threadIdx.x & 31;
    const int b    = blockIdx.x / (CC / 128);
    const int col0 = (blockIdx.x % (CC / 128)) * 128 + wave * 16;   // channel
    const int m    = lane & 15, g = lane >> 4;

    const _Float16* arow = probs + ((size_t)b * TT + m) * NN;
    const float*    bcol = x + ((size_t)b * CC + col0 + m) * NN + g * 16;

    v8f acc = {};
    for (int k0 = 0; k0 < NN; k0 += 32) {
        __builtin_prefetch(bcol + k0 + 256, 0, 0);
        v16h a  = load_fragA(arow, k0, g);
        v16h bf = load_fragB_f32(bcol, k0);
        acc = __builtin_amdgcn_wmma_f32_16x16x32_f16(false, a, false, bf,
                                                     (short)0, acc, false, false);
    }
#pragma unroll
    for (int r = 0; r < 8; ++r) {
        int row = r + g * 8;
        yh[((size_t)b * TT + row) * CC + col0 + m] = (_Float16)acc[r];
    }
}

// ---------------------------------------------------------------------------
// K6: 1x1 conv GEMM + BatchNorm:  Z[t,o] = BN(sum_c Y[t,c]*W[o,c] + cb[o])
//     grid = B * (C/128), block = 256.
// ---------------------------------------------------------------------------
__global__ void k_conv_bn(const _Float16* __restrict__ yh,
                          const float* __restrict__ w,
                          const float* __restrict__ cb,
                          const float* __restrict__ gamma,
                          const float* __restrict__ beta,
                          const float* __restrict__ mean,
                          const float* __restrict__ var,
                          float* __restrict__ ybn) {
    const int wave = threadIdx.x >> 5;
    const int lane = threadIdx.x & 31;
    const int b    = blockIdx.x / (CC / 128);
    const int col0 = (blockIdx.x % (CC / 128)) * 128 + wave * 16;   // out channel o
    const int m    = lane & 15, g = lane >> 4;
    const int o    = col0 + m;

    const _Float16* arow = yh + ((size_t)b * TT + m) * CC;
    const float*    bcol = w + (size_t)o * CC + g * 16;

    v8f acc = {};
    for (int k0 = 0; k0 < CC; k0 += 32) {
        __builtin_prefetch(bcol + k0 + 256, 0, 0);
        v16h a  = load_fragA(arow, k0, g);
        v16h bf = load_fragB_f32(bcol, k0);
        acc = __builtin_amdgcn_wmma_f32_16x16x32_f16(false, a, false, bf,
                                                     (short)0, acc, false, false);
    }
    const float sc = gamma[o] * rsqrtf(var[o] + EPS_BN);
    const float bs = cb[o];
    const float mu = mean[o];
    const float bt = beta[o];
#pragma unroll
    for (int r = 0; r < 8; ++r) {
        int row = r + g * 8;
        ybn[((size_t)b * TT + row) * CC + o] = (acc[r] + bs - mu) * sc + bt;
    }
}

// ---------------------------------------------------------------------------
// K7: residual broadcast:  z[b,c,t,p] = x[b,c,t,p] + ybn[b,t,c]   (float4)
// ---------------------------------------------------------------------------
__global__ void k_residual(const float* __restrict__ x,
                           const float* __restrict__ ybn,
                           float* __restrict__ out) {
    const size_t idx = (size_t)blockIdx.x * 256 + threadIdx.x;   // float4 index
    const int t  = (int)((idx >> 5) & 15);
    const size_t cb = idx >> 9;
    const int c  = (int)(cb & 2047);
    const int b  = (int)(cb >> 11);

    float4 xv = ((const float4*)x)[idx];
    float  s  = ybn[((size_t)b * TT + t) * CC + c];
    float4 ov = { xv.x + s, xv.y + s, xv.z + s, xv.w + s };
    ((float4*)out)[idx] = ov;
}

// ---------------------------------------------------------------------------
extern "C" void kernel_launch(void* const* d_in, const int* in_sizes, int n_in,
                              void* d_out, int out_size, void* d_ws, size_t ws_size,
                              hipStream_t stream) {
    (void)in_sizes; (void)n_in; (void)out_size; (void)ws_size;

    const float* x     = (const float*)d_in[0];
    const float* cw    = (const float*)d_in[1];
    const float* cb    = (const float*)d_in[2];
    const float* gamma = (const float*)d_in[3];
    const float* beta  = (const float*)d_in[4];
    const float* mean  = (const float*)d_in[5];
    const float* var   = (const float*)d_in[6];
    float* out = (float*)d_out;

    // workspace layout (bytes, 256B aligned)
    char* ws = (char*)d_ws;
    size_t off = 0;
    _Float16* mn_h = (_Float16*)(ws + off); off += (size_t)BB * NN * CC * 2;        // 128 MiB
    _Float16* qn_h = (_Float16*)(ws + off); off += (size_t)BB * TT * CC * 2;        //   1 MiB
    float*   scores = (float*)  (ws + off); off += (size_t)BB * TT * NN * 4;        //   2 MiB
    _Float16* probs = (_Float16*)(ws + off); off += (size_t)BB * TT * NN * 2;       //   1 MiB
    _Float16* yh   = (_Float16*)(ws + off); off += (size_t)BB * TT * CC * 2;        //   1 MiB
    float*   ybn   = (float*)   (ws + off); off += (size_t)BB * TT * CC * 4;        //   2 MiB
    float*   n2p   = (float*)   (ws + off); off += (size_t)CSEG * BB * NN * 4;      //   1 MiB

    // 1) pool -> mem (f16, (b,n,c)) + per-segment column norms (no atomics)
    k_pool<<<dim3(BB * TT, CSEG), 256, 0, stream>>>(x, mn_h, n2p);
    // 2) query means (f16, (b,t,c))
    k_query<<<(BB * TT * CC) / 8, 256, 0, stream>>>(x, qn_h);
    // 3) raw cosine-numerator GEMM (WMMA f16 -> f32)
    k_score<<<BB * (NN / 128), 256, 0, stream>>>(qn_h, mn_h, scores);
    // 4) rescale by inv-norms, mask, softmax
    k_softmax<<<BB * TT, 256, 0, stream>>>(scores, qn_h, n2p, probs);
    // 5) attention aggregation GEMM (WMMA, x converted f32->f16 in-register)
    k_agg<<<BB * (CC / 128), 256, 0, stream>>>(probs, x, yh);
    // 6) 1x1 conv GEMM + BN (WMMA)
    k_conv_bn<<<BB * (CC / 128), 256, 0, stream>>>(yh, cw, cb, gamma, beta, mean, var, ybn);
    // 7) residual broadcast add
    k_residual<<<(BB * CC * TT * HWV / 4) / 256, 256, 0, stream>>>(x, ybn, out);
}
